// tokenMixer_83777632076022
// MI455X (gfx1250) — compile-verified, matrix operand order checked by reading, not run on hardware
//
#include <hip/hip_runtime.h>
#include <hip/hip_bf16.h>
#include <cmath>

typedef __bf16 bf16_t;
typedef __attribute__((ext_vector_type(16))) __bf16 v16bf;
typedef __attribute__((ext_vector_type(8)))  __bf16 v8bf;
typedef __attribute__((ext_vector_type(4)))  __bf16 v4bf;
typedef __attribute__((ext_vector_type(8)))  float  v8f;
typedef __attribute__((ext_vector_type(4)))  float  v4f;

#define BB   4
#define LSEQ 8192
#define DD   512
#define HH   1024
#define BLR  32768      // B*L rows
#define NCH  64         // chunks per sequence
#define CL   128        // chunk length (NCH*CL == LSEQ)

// ---------------- math helpers ----------------
__device__ __forceinline__ float softplusf(float x) {
    return fmaxf(x, 0.f) + log1pf(__expf(-fabsf(x)));
}
__device__ __forceinline__ float logaddexpf_(float a, float b) {
    float m = fmaxf(a, b);
    if (m < -1e30f) return -__builtin_inff();
    return m + log1pf(__expf(fminf(a, b) - m));
}

// ---------------- CDNA5 async global->LDS copy ----------------
__device__ __forceinline__ void async_copy_b128(uint32_t lds_byte, const void* g) {
    asm volatile("global_load_async_to_lds_b128 %0, %1, off"
                 :: "v"(lds_byte), "v"((unsigned long long)(uintptr_t)g)
                 : "memory");
}
__device__ __forceinline__ void wait_async6() {
    asm volatile("s_wait_asynccnt 0x6" ::: "memory");
}
__device__ __forceinline__ void wait_async0() {
    asm volatile("s_wait_asynccnt 0x0" ::: "memory");
}

// ---------------- fp32 -> bf16 conversion ----------------
__global__ __launch_bounds__(256) void cvt_f32_bf16(
    const float* __restrict__ in, bf16_t* __restrict__ out, long n)
{
    size_t i = ((size_t)blockIdx.x * 256 + threadIdx.x) * 4;
    if (i >= (size_t)n) return;
    v4f f = *(const v4f*)(in + i);
    v4bf o;
    #pragma unroll
    for (int j = 0; j < 4; j++) o[j] = (bf16_t)f[j];
    *(v4bf*)(out + i) = o;
}

// ---------------- bf16 WMMA GEMM, async double-buffered staging ----------------
// C[M x N] = A[M x K] * Bw[N x K]^T (+ bias). A, Bw bf16 row-major.
// Block tile 256(M) x 128(N), 8 waves in 4x2 grid, 64x64 per wave (4x4 subtiles),
// BK=32: 16 ds_load_b128 -> 16 v_wmma per step (32 FLOP / LDS byte).
#define BM 256
#define BN 128
#define BK 32
#define LDSTR 40   // padded bf16 row stride (80B, 16B-aligned rows)

template <bool HASB>
__global__ __launch_bounds__(256) void gemm_wmma_bf16(
    const bf16_t* __restrict__ A, const bf16_t* __restrict__ Bw,
    const float* __restrict__ bias, float* __restrict__ C,
    int N, int K)
{
    __shared__ __align__(16) bf16_t As[2][BM * LDSTR];   // 40 KB
    __shared__ __align__(16) bf16_t Bs[2][BN * LDSTR];   // 20 KB

    const int tid  = threadIdx.x;
    const int row0 = blockIdx.y * BM;
    const int col0 = blockIdx.x * BN;
    const int wave = tid >> 5, lane = tid & 31;
    const int wm   = wave >> 1, wn = wave & 1;      // 4x2 wave grid
    const int fr   = lane & 15, half = lane >> 4;   // fragment row / K-half

    v8f acc[4][4];
    #pragma unroll
    for (int i = 0; i < 4; i++)
        #pragma unroll
        for (int j = 0; j < 4; j++) acc[i][j] = v8f{};

    // per-thread staging: A one row of 32 (4x b128), B half row of 16 (2x b128)
    const int br = tid >> 1, bc = (tid & 1) << 4;

    const bf16_t* gA = A  + (size_t)(row0 + tid) * K;
    const bf16_t* gB = Bw + (size_t)(col0 + br) * K + bc;
    const uint32_t lA[2] = { (uint32_t)(uintptr_t)&As[0][tid * LDSTR],
                             (uint32_t)(uintptr_t)&As[1][tid * LDSTR] };
    const uint32_t lB[2] = { (uint32_t)(uintptr_t)&Bs[0][br * LDSTR + bc],
                             (uint32_t)(uintptr_t)&Bs[1][br * LDSTR + bc] };

    auto stage = [&](int k, int buf) {
        const bf16_t* a = gA + (size_t)k * BK;
        const bf16_t* b = gB + (size_t)k * BK;
        async_copy_b128(lA[buf],      a);
        async_copy_b128(lA[buf] + 16, a + 8);
        async_copy_b128(lA[buf] + 32, a + 16);
        async_copy_b128(lA[buf] + 48, a + 24);
        async_copy_b128(lB[buf],      b);
        async_copy_b128(lB[buf] + 16, b + 8);
    };

    auto compute = [&](int cur) {
        v16bf af[4], bfr[4];
        #pragma unroll
        for (int mi = 0; mi < 4; mi++) {
            const bf16_t* p = &As[cur][(wm * 64 + mi * 16 + fr) * LDSTR + half * 8];
            v8bf lo = *(const v8bf*)p;
            v8bf hi = *(const v8bf*)(p + 16);
            #pragma unroll
            for (int i = 0; i < 8; i++) { af[mi][i] = lo[i]; af[mi][8 + i] = hi[i]; }
        }
        #pragma unroll
        for (int ni = 0; ni < 4; ni++) {
            const bf16_t* p = &Bs[cur][(wn * 64 + ni * 16 + fr) * LDSTR + half * 8];
            v8bf lo = *(const v8bf*)p;
            v8bf hi = *(const v8bf*)(p + 16);
            #pragma unroll
            for (int i = 0; i < 8; i++) { bfr[ni][i] = lo[i]; bfr[ni][8 + i] = hi[i]; }
        }
        #pragma unroll
        for (int mi = 0; mi < 4; mi++)
            #pragma unroll
            for (int ni = 0; ni < 4; ni++)
                acc[mi][ni] = __builtin_amdgcn_wmma_f32_16x16x32_bf16(
                    false, af[mi], false, bfr[ni],
                    (short)0, acc[mi][ni], false, false);
    };

    const int steps = K / BK;

    stage(0, 0);                       // prologue
    #pragma unroll 2
    for (int k = 0; k < steps - 1; k++) {
        stage(k + 1, (k + 1) & 1);     // overlap next tile's copy
        wait_async6();                 // in-order: current tile's 6 copies done
        __syncthreads();
        compute(k & 1);
        __syncthreads();               // all reads done before this buf restaged
    }
    wait_async0();                     // last tile
    __syncthreads();
    compute((steps - 1) & 1);

    // epilogue: C/D layout — VGPR r holds M = r + 8*half, N = lane&15
    #pragma unroll
    for (int mi = 0; mi < 4; mi++) {
        #pragma unroll
        for (int ni = 0; ni < 4; ni++) {
            const int col  = col0 + wn * 64 + ni * 16 + fr;
            const float bv = HASB ? bias[col] : 0.f;
            const int rowb = row0 + wm * 64 + mi * 16 + half * 8;
            float* cp = C + (size_t)rowb * N + col;
            #pragma unroll
            for (int r = 0; r < 8; r++)
                cp[(size_t)r * N] = acc[mi][ni][r] + bv;
        }
    }
}

// ---------------- pointwise: hg -> (lc, s) in place ----------------
__global__ __launch_bounds__(256) void pointwise_kernel(
    float* __restrict__ hg, const float* __restrict__ b_in)
{
    size_t idx = (size_t)blockIdx.x * 256 + threadIdx.x;   // over BLR*HH
    size_t row = idx >> 10;
    int h = (int)(idx & (HH - 1));
    float* ph = hg + row * (size_t)(2 * HH) + h;
    float* pg = ph + HH;
    float hidden = *ph + b_in[h];
    float gate   = *pg + b_in[HH + h];
    float lc  = -softplusf(gate);            // log_coeffs
    float lz  = -softplusf(-gate);           // log_z
    float lth = (hidden >= 0.f) ? __logf(hidden + 0.5f) : -softplusf(-hidden);
    *ph = lc;
    *pg = lz + lth;                          // s
}

// ---------------- scan pass 1: per-chunk aggregates ----------------
__global__ __launch_bounds__(256) void scan_chunk_kernel(
    const float* __restrict__ hg, float* __restrict__ Ab, float* __restrict__ Sb)
{
    int h  = blockIdx.y * 256 + threadIdx.x;
    int bc = blockIdx.x;                     // b*NCH + chunk
    int b  = bc >> 6, c = bc & (NCH - 1);
    const float* base = hg + (size_t)(b * LSEQ + c * CL) * (2 * HH) + h;
    float A = 0.f, S = -__builtin_inff();
    for (int t = 0; t < CL; t++) {
        float lc = base[(size_t)t * (2 * HH)];
        float s  = base[(size_t)t * (2 * HH) + HH];
        S = logaddexpf_(S + lc, s);
        A += lc;
    }
    size_t o = (size_t)bc * HH + h;
    Ab[o] = A; Sb[o] = S;
}

// ---------------- scan pass 2: sequential carry over chunks ----------------
__global__ __launch_bounds__(256) void scan_carry_kernel(
    const float* __restrict__ Ab, const float* __restrict__ Sb, float* __restrict__ Cb)
{
    int idx = blockIdx.x * 256 + threadIdx.x;  // b*HH + h
    int b = idx >> 10, h = idx & (HH - 1);
    float c = -__builtin_inff();
    for (int k = 0; k < NCH; k++) {
        size_t o = (size_t)(b * NCH + k) * HH + h;
        Cb[o] = c;
        c = logaddexpf_(c + Ab[o], Sb[o]);
    }
}

// ---------------- scan pass 3: apply + exp -> bf16 h ----------------
__global__ __launch_bounds__(256) void scan_apply_kernel(
    const float* __restrict__ hg, const float* __restrict__ Cb, bf16_t* __restrict__ hout)
{
    int h  = blockIdx.y * 256 + threadIdx.x;
    int bc = blockIdx.x;
    int b  = bc >> 6, c = bc & (NCH - 1);
    const float* base = hg + (size_t)(b * LSEQ + c * CL) * (2 * HH) + h;
    bf16_t* ob = hout + (size_t)(b * LSEQ + c * CL) * HH + h;
    float lh = Cb[(size_t)bc * HH + h];
    for (int t = 0; t < CL; t++) {
        float lc = base[(size_t)t * (2 * HH)];
        float s  = base[(size_t)t * (2 * HH) + HH];
        lh = logaddexpf_(lc + lh, s);
        ob[(size_t)t * HH] = (bf16_t)__expf(lh);
    }
}

// ---------------- launch ----------------
extern "C" void kernel_launch(void* const* d_in, const int* in_sizes, int n_in,
                              void* d_out, int out_size, void* d_ws, size_t ws_size,
                              hipStream_t stream)
{
    const float* x     = (const float*)d_in[0];
    const float* W_in  = (const float*)d_in[1];
    const float* b_in  = (const float*)d_in[2];
    const float* W_out = (const float*)d_in[3];
    const float* b_out = (const float*)d_in[4];
    float* out = (float*)d_out;

    char* ws = (char*)d_ws;
    float*  hg    = (float*)ws;                                 // BLR*2H fp32 (256MB)
    size_t  off   = (size_t)BLR * 2 * HH * sizeof(float);
    bf16_t* hbuf  = (bf16_t*)(ws + off);  off += (size_t)BLR * HH * sizeof(bf16_t);
    float*  Ab    = (float*)(ws + off);   off += (size_t)BB * NCH * HH * sizeof(float);
    float*  Sb    = (float*)(ws + off);   off += (size_t)BB * NCH * HH * sizeof(float);
    float*  Cb    = (float*)(ws + off);   off += (size_t)BB * NCH * HH * sizeof(float);
    bf16_t* xb    = (bf16_t*)(ws + off);  off += (size_t)BLR * DD * sizeof(bf16_t);
    bf16_t* Winb  = (bf16_t*)(ws + off);  off += (size_t)2 * HH * DD * sizeof(bf16_t);
    bf16_t* Woutb = (bf16_t*)(ws + off);

    // one-shot fp32 -> bf16 conversions
    cvt_f32_bf16<<<(BLR * DD) / (4 * 256), 256, 0, stream>>>(x, xb, (long)BLR * DD);
    cvt_f32_bf16<<<(2 * HH * DD) / (4 * 256), 256, 0, stream>>>(W_in, Winb, (long)2 * HH * DD);
    cvt_f32_bf16<<<(DD * HH) / (4 * 256), 256, 0, stream>>>(W_out, Woutb, (long)DD * HH);

    // GEMM1: hg = x @ W_in^T   (M=32768, N=2048, K=512)
    gemm_wmma_bf16<false><<<dim3((2 * HH) / BN, BLR / BM), 256, 0, stream>>>(
        xb, Winb, nullptr, hg, 2 * HH, DD);

    // activations -> (lc, s) in place
    pointwise_kernel<<<(BLR * HH) / 256, 256, 0, stream>>>(hg, b_in);

    // chunked associative scan over L
    scan_chunk_kernel<<<dim3(BB * NCH, HH / 256), 256, 0, stream>>>(hg, Ab, Sb);
    scan_carry_kernel<<<(BB * HH) / 256, 256, 0, stream>>>(Ab, Sb, Cb);
    scan_apply_kernel<<<dim3(BB * NCH, HH / 256), 256, 0, stream>>>(hg, Cb, hbuf);

    // GEMM2: out = h @ W_out^T + b_out   (M=32768, N=512, K=1024)
    gemm_wmma_bf16<true><<<dim3(DD / BN, BLR / BM), 256, 0, stream>>>(
        hbuf, Woutb, b_out, out, DD, HH);
}
